// CrossAttention_28063316312219
// MI455X (gfx1250) — matrix operand order repaired and checked by reading, not compiled, not run on hardware
//
#include <hip/hip_runtime.h>

// CrossAttention for MI455X (gfx1250). All GEMM stages (conv-as-implicit-GEMM,
// qkv, attention scores, PV, proj) run on V_WMMA_F32_16X16X32_BF16 with f32
// accumulation. Tile staging uses the Tensor Data Mover (tensor_load_to_lds +
// s_wait_tensorcnt) when the toolchain exposes it, else 128-bit vector loads.
// Workspace requirement: ~156 MiB.

typedef __bf16 bf16_t;
typedef __attribute__((ext_vector_type(16))) __bf16 v16bf;
typedef __attribute__((ext_vector_type(8)))  __bf16 v8bf;
typedef __attribute__((ext_vector_type(8)))  float  v8f;
typedef __attribute__((ext_vector_type(4)))  unsigned int u32x4;
typedef __attribute__((ext_vector_type(4)))  int i32x4;
typedef __attribute__((ext_vector_type(8)))  int i32x8;

#if defined(__HIP_DEVICE_COMPILE__) && __has_builtin(__builtin_amdgcn_tensor_load_to_lds) && \
    __has_builtin(__builtin_amdgcn_s_wait_tensorcnt)
#define HAVE_TDM 1
#else
#define HAVE_TDM 0
#endif

static __device__ __forceinline__ v8f wmma_bf16(v16bf a, v16bf b, v8f c) {
  // 8 args: (neg_a, A, neg_b, B, c_mod, C, reuse_a, reuse_b)
  return __builtin_amdgcn_wmma_f32_16x16x32_bf16(false, a, false, b, (short)0, c,
                                                 false, false);
}

#if HAVE_TDM
// Issue one TDM 2-D tile load: tile_d1 rows of tile_d0 bf16 elements, source
// row pitch `pitch` elements, destination LDS rows packed contiguously.
// D# layout per CDNA5 ISA ch.8 (group0: count/lds/global/type, group1: dims).
static __device__ __forceinline__ void tdm_load_2d(const void* gsrc,
                                                   unsigned lds_off,
                                                   unsigned tile_d0,
                                                   unsigned tile_d1,
                                                   unsigned tensor_d0,
                                                   unsigned tensor_d1,
                                                   unsigned pitch) {
  unsigned long long ga = (unsigned long long)gsrc;
  u32x4 g0;
  g0[0] = 1u;                                    // count=1, user descriptor
  g0[1] = lds_off;                               // lds_addr (bytes)
  g0[2] = (unsigned)(ga & 0xffffffffu);          // global_addr[31:0]
  g0[3] = (unsigned)((ga >> 32) & 0x01ffffffu)   // global_addr[56:32]
        | (2u << 30);                            // type = 2 ("image")
  i32x8 g1;
  g1[0] = 1 << 16;                               // data_size = 2 bytes
  g1[1] = (int)(tensor_d0 << 16);                // tensor_dim0[15:0]
  g1[2] = (int)((tensor_d0 >> 16) | (tensor_d1 << 16));
  g1[3] = (int)((tensor_d1 >> 16) | (tile_d0 << 16));   // tile_dim0
  g1[4] = (int)(tile_d1 & 0xffffu);              // tile_dim1 (tile_dim2 = 0)
  g1[5] = (int)pitch;                            // tensor_dim0_stride[31:0]
  g1[6] = 0;
  g1[7] = 0;
  i32x4 z4 = {0, 0, 0, 0};
#if __clang_major__ >= 23
  i32x8 z8 = {0, 0, 0, 0, 0, 0, 0, 0};
  __builtin_amdgcn_tensor_load_to_lds(g0, g1, z4, z4, z8, 0);
#else
  __builtin_amdgcn_tensor_load_to_lds(g0, g1, z4, z4, 0);
#endif
}

static __device__ __forceinline__ unsigned lds_offset_of(const void* p) {
  return (unsigned)(size_t)p;  // low 32 bits of generic LDS address = offset
}
#endif

// A-matrix 16x32 (MxK) fragment from row-major [m][k] LDS, leading dim ld.
// Contiguous k order used here (same as B loaders): lanes 0-15 -> M=lane, K=0..15; lanes 16-31 -> K=16..31.
static __device__ __forceinline__ v16bf load_frag_a(const bf16_t* s, int ld) {
  const int lane = threadIdx.x & 31;
  const int row  = lane & 15;
  const int kb   = (lane >> 4) << 4;
  v16bf a;
#pragma unroll
  for (int j = 0; j < 8; ++j) {
    a[j]     = s[row * ld + kb + j];
    a[j + 8] = s[row * ld + kb + 8 + j];
  }
  return a;
}

// A fragment for the conv halo: Ha is [(4 rows)*(34 cols)][32 k], row index is
// derived from this lane's spatial position shifted by the tap (dy,dx).
static __device__ __forceinline__ v16bf load_frag_a_halo(const bf16_t (*Ha)[32],
                                                         int pbase, int dy, int dx) {
  const int lane = threadIdx.x & 31;
  const int p = pbase + (lane & 15);
  const int y = p >> 5;
  const int x = p & 31;
  const bf16_t* s = Ha[(y + dy + 1) * 34 + (x + dx + 1)];
  const int kb = (lane >> 4) << 4;
  v16bf a;
#pragma unroll
  for (int j = 0; j < 8; ++j) {
    a[j]     = s[kb + j];
    a[j + 8] = s[kb + 8 + j];
  }
  return a;
}

// B-matrix 32x16 (KxN) fragment, source stored [n][k] (row = N, ld elems).
static __device__ __forceinline__ v16bf load_frag_b_nk(const bf16_t* s, int ld) {
  const int lane = threadIdx.x & 31;
  const int n  = lane & 15;
  const int kb = (lane >> 4) << 4;
  v16bf b;
#pragma unroll
  for (int j = 0; j < 16; ++j) b[j] = s[n * ld + kb + j];
  return b;
}

// B-matrix 32x16 (KxN) fragment, source stored [k][n] (row = K, ld elems).
static __device__ __forceinline__ v16bf load_frag_b_kn(const bf16_t* s, int ld) {
  const int lane = threadIdx.x & 31;
  const int n  = lane & 15;
  const int kb = (lane >> 4) << 4;
  v16bf b;
#pragma unroll
  for (int j = 0; j < 16; ++j) b[j] = s[(kb + j) * ld + n];
  return b;
}

// ---------------------------------------------------------------- converts
__global__ void k_f32_to_bf16(const float* __restrict__ src,
                              bf16_t* __restrict__ dst, int n8) {
  int i = blockIdx.x * blockDim.x + threadIdx.x;
  int stride = gridDim.x * blockDim.x;
  for (; i < n8; i += stride) {
    v8f f = ((const v8f*)src)[i];
    v8bf o;
#pragma unroll
    for (int j = 0; j < 8; ++j) o[j] = (bf16_t)f[j];
    ((v8bf*)dst)[i] = o;
  }
}

// conv_w [1024][2048][3][3] -> wt [9][1024][2048] bf16 (branch/div free)
__global__ void k_repack_convw(const float* __restrict__ w,
                               bf16_t* __restrict__ wt) {
  int i = blockIdx.x * blockDim.x + threadIdx.x;
  int stride = gridDim.x * blockDim.x;
  const int total = 9 << 21;
  for (; i < total; i += stride) {
    int t   = i >> 21;              // tap
    int rem = i & ((1 << 21) - 1);  // oc*2048 + ic
    wt[i] = (bf16_t)w[rem * 9 + t];
  }
}

// ------------------------------------------------------------- conv (GEMM)
// out[b][oc][p] = sum_t sum_ic wt[t][oc][ic] * x{1,2}[b][ic&1023][p shift t]
// Halo tile (4 rows x 34 cols x 32 ch) loaded once per K-slice, reused by all
// 9 taps; 9 weight tiles staged per K-slice.
__global__ __launch_bounds__(256) void k_conv(const bf16_t* __restrict__ x1b,
                                              const bf16_t* __restrict__ x2b,
                                              const bf16_t* __restrict__ wt,
                                              const float* __restrict__ bias,
                                              bf16_t* __restrict__ xcat) {
  __shared__ __align__(16) bf16_t Ha[4 * 34][32];
  __shared__ __align__(16) bf16_t Bs[9][64][32];
  const int tid  = threadIdx.x;
  const int b    = blockIdx.z;
  const int p0   = blockIdx.y * 64;   // two full 32-wide image rows
  const int y0   = p0 >> 5;
  const int n0   = blockIdx.x * 64;   // output-channel tile
  const int wv   = tid >> 5;
  const int lane = tid & 31;
  const int mi   = wv & 3;
  const int nj0  = (wv >> 2) * 2;

  // zero halo borders (x = -1 and x = 32); never touched by the main fill
  {
    int ry = tid >> 6, side = (tid >> 5) & 1, kk = tid & 31;
    Ha[ry * 34 + side * 33][kk] = (bf16_t)0.0f;
  }

  v8f acc0 = {};
  v8f acc1 = {};

  const int kk  = tid & 31;
  const int grp = tid >> 5;           // uniform per wave
  const int ry  = grp >> 1;           // 0..3
  const int xh  = (grp & 1) * 16;     // 0 or 16
  const int gy  = y0 + ry - 1;
  const bool rowok = (gy >= 0) && (gy < 32);

  const int bn  = tid >> 2;           // 0..63
  const int bkb = (tid & 3) * 8;

  for (int k0 = 0; k0 < 2048; k0 += 32) {
    const bf16_t* src = (k0 < 1024) ? x1b : x2b;
    const bf16_t* srow = src + (((size_t)(b * 1024 + (k0 & 1023) + kk)) << 10);
#pragma unroll
    for (int ii = 0; ii < 2; ++ii) {
      int x8 = xh + ii * 8;
      v8bf val = {};
      if (rowok) val = *(const v8bf*)(srow + gy * 32 + x8);
#pragma unroll
      for (int e = 0; e < 8; ++e) Ha[ry * 34 + x8 + e + 1][kk] = val[e];
    }
#pragma unroll
    for (int t = 0; t < 9; ++t) {
      *(v8bf*)&Bs[t][bn][bkb] =
          *(const v8bf*)(wt + ((size_t)t * 1024 + n0 + bn) * 2048 + k0 + bkb);
    }
    __syncthreads();
#pragma unroll
    for (int t = 0; t < 9; ++t) {
      const int dy = t / 3 - 1;
      const int dx = t % 3 - 1;
      v16bf a  = load_frag_a_halo(Ha, mi * 16, dy, dx);
      v16bf b0 = load_frag_b_nk(&Bs[t][(nj0 + 0) * 16][0], 32);
      v16bf b1 = load_frag_b_nk(&Bs[t][(nj0 + 1) * 16][0], 32);
      acc0 = wmma_bf16(a, b0, acc0);
      acc1 = wmma_bf16(a, b1, acc1);
    }
    __syncthreads();
  }

  // epilogue: contiguous along r -> one 128-bit store per accumulator
  const int pgb = p0 + mi * 16 + ((lane >> 4) << 3);
  const int oc0 = n0 + (nj0 + 0) * 16 + (lane & 15);
  const int oc1 = n0 + (nj0 + 1) * 16 + (lane & 15);
  v8bf pk0, pk1;
#pragma unroll
  for (int r = 0; r < 8; ++r) {
    pk0[r] = (bf16_t)(acc0[r] + bias[oc0]);
    pk1[r] = (bf16_t)(acc1[r] + bias[oc1]);
  }
  *(v8bf*)&xcat[(((size_t)(b * 1024 + oc0)) << 10) + pgb] = pk0;
  *(v8bf*)&xcat[(((size_t)(b * 1024 + oc1)) << 10) + pgb] = pk1;
}

// --------------------------------------------------------------- qkv GEMM
// C[m][n] = scale * sum_k A[m][k] * W[n][k]; 64x128 tile; scatter to
// [b][h][tok][d] bf16. A/B tiles staged by the TDM when available.
__global__ __launch_bounds__(256) void k_gemm_qkv(const bf16_t* __restrict__ A,
                                                  const bf16_t* __restrict__ W,
                                                  bf16_t* __restrict__ dst,
                                                  float scale) {
  __shared__ __align__(16) bf16_t As[64][32];
  __shared__ __align__(16) bf16_t Bs[128][32];
  const int tid  = threadIdx.x;
  const int m0   = blockIdx.y * 64;
  const int n0   = blockIdx.x * 128;
  const int wv   = tid >> 5;
  const int lane = tid & 31;
  const int mi   = wv & 3;
  const int nj0  = (wv >> 2) * 4;

  v8f acc[4] = {};

  const int lr  = tid >> 2;        // 0..63
  const int lkb = (tid & 3) * 8;

  for (int k0 = 0; k0 < 1024; k0 += 32) {
#if HAVE_TDM
    if (wv == 0) {
      tdm_load_2d(&A[(size_t)m0 * 1024 + k0], lds_offset_of(&As[0][0]),
                  32, 64, 1024, 8192, 1024);
      tdm_load_2d(&W[(size_t)n0 * 1024 + k0], lds_offset_of(&Bs[0][0]),
                  32, 128, 1024, 1024, 1024);
      __builtin_amdgcn_s_wait_tensorcnt(0);
    }
#else
    *(v8bf*)&As[lr][lkb] = *(const v8bf*)&A[(size_t)(m0 + lr) * 1024 + k0 + lkb];
#pragma unroll
    for (int jj = 0; jj < 2; ++jj) {
      int n = lr + 64 * jj;
      *(v8bf*)&Bs[n][lkb] = *(const v8bf*)&W[(size_t)(n0 + n) * 1024 + k0 + lkb];
    }
    if (k0 + 32 < 1024) {
      __builtin_prefetch(&A[(size_t)(m0 + lr) * 1024 + k0 + 32 + lkb], 0, 0);
      __builtin_prefetch(&W[(size_t)(n0 + lr) * 1024 + k0 + 32 + lkb], 0, 0);
    }
#endif
    __syncthreads();
    v16bf a = load_frag_a(&As[mi * 16][0], 32);
#pragma unroll
    for (int jj = 0; jj < 4; ++jj) {
      v16bf bf = load_frag_b_nk(&Bs[(nj0 + jj) * 16][0], 32);
      acc[jj] = wmma_bf16(a, bf, acc[jj]);
    }
    __syncthreads();
  }
#pragma unroll
  for (int r = 0; r < 8; ++r) {
    int m = m0 + mi * 16 + ((lane >> 4) << 3) + r;
    int bi = m >> 10;
    int tok = m & 1023;
#pragma unroll
    for (int jj = 0; jj < 4; ++jj) {
      int c = n0 + (nj0 + jj) * 16 + (lane & 15);
      int h = c >> 7;
      int d = c & 127;
      dst[(((size_t)(bi * 8 + h) * 1024 + tok) << 7) + d] =
          (bf16_t)(acc[jj][r] * scale);
    }
  }
}

// --------------------------------------------------------------- attention
// Flash-style: one block per (b, h, 64 queries). q/k/v: [b][h][1024][128] bf16.
__global__ __launch_bounds__(256) void k_attention(const bf16_t* __restrict__ q,
                                                   const bf16_t* __restrict__ k,
                                                   const bf16_t* __restrict__ v,
                                                   bf16_t* __restrict__ attnb) {
  __shared__ __align__(16) bf16_t Qs[64][128];
  __shared__ __align__(16) bf16_t Ks[64][128];
  __shared__ __align__(16) bf16_t Vs[64][128];
  __shared__ __align__(16) bf16_t Ps[8][16][64];  // per-wave P relayout buffer
  const int tid   = threadIdx.x;
  const int b     = blockIdx.z;
  const int h     = blockIdx.y;
  const int bh    = b * 8 + h;
  const int q0    = blockIdx.x * 64;
  const int wv    = tid >> 5;
  const int lane  = tid & 31;
  const int mi    = wv & 3;    // 16-row query sub-tile
  const int dhalf = wv >> 2;   // which 64 of the 128 head dims

  const bf16_t* qp = q + (size_t)(bh * 1024 + q0) * 128;
#pragma unroll
  for (int i = 0; i < 4; ++i) {
    int e = (tid + 256 * i) << 3;
    *(v8bf*)&Qs[e >> 7][e & 127] = *(const v8bf*)&qp[e];
  }

  float rmax[8], rsum[8];
  v8f o0 = {}, o1 = {}, o2 = {}, o3 = {};
#pragma unroll
  for (int r = 0; r < 8; ++r) { rmax[r] = -1e30f; rsum[r] = 0.0f; }

  for (int jb = 0; jb < 16; ++jb) {
    const bf16_t* kp = k + (size_t)(bh * 1024 + jb * 64) * 128;
    const bf16_t* vp = v + (size_t)(bh * 1024 + jb * 64) * 128;
    __syncthreads();
#if HAVE_TDM
    if (wv == 0) {
      tdm_load_2d(kp, lds_offset_of(&Ks[0][0]), 128, 64, 128, 1024, 128);
      tdm_load_2d(vp, lds_offset_of(&Vs[0][0]), 128, 64, 128, 1024, 128);
      __builtin_amdgcn_s_wait_tensorcnt(0);
    }
#else
#pragma unroll
    for (int i = 0; i < 4; ++i) {
      int e = (tid + 256 * i) << 3;
      *(v8bf*)&Ks[e >> 7][e & 127] = *(const v8bf*)&kp[e];
      *(v8bf*)&Vs[e >> 7][e & 127] = *(const v8bf*)&vp[e];
    }
#endif
    __syncthreads();

    // S[16 x 64] for this wave's query rows over the 64 keys of the block.
    v8f s0 = {}, s1 = {}, s2 = {}, s3 = {};
#pragma unroll
    for (int ks = 0; ks < 4; ++ks) {
      v16bf a = load_frag_a(&Qs[mi * 16][ks * 32], 128);
      s0 = wmma_bf16(a, load_frag_b_nk(&Ks[0][ks * 32], 128), s0);
      s1 = wmma_bf16(a, load_frag_b_nk(&Ks[16][ks * 32], 128), s1);
      s2 = wmma_bf16(a, load_frag_b_nk(&Ks[32][ks * 32], 128), s2);
      s3 = wmma_bf16(a, load_frag_b_nk(&Ks[48][ks * 32], 128), s3);
    }

    // Online softmax. C/D layout: col = lane%16, row = (lane/16)*8 + r, so a
    // full row lives in one 16-lane half -> xor-reductions over {1,2,4,8}.
#pragma unroll
    for (int r = 0; r < 8; ++r) {
      float tmax = fmaxf(fmaxf(s0[r], s1[r]), fmaxf(s2[r], s3[r]));
      tmax = fmaxf(tmax, __shfl_xor(tmax, 1));
      tmax = fmaxf(tmax, __shfl_xor(tmax, 2));
      tmax = fmaxf(tmax, __shfl_xor(tmax, 4));
      tmax = fmaxf(tmax, __shfl_xor(tmax, 8));
      float nm   = fmaxf(rmax[r], tmax);
      float corr = __expf(rmax[r] - nm);
      rmax[r] = nm;
      float p0v = __expf(s0[r] - nm);
      float p1v = __expf(s1[r] - nm);
      float p2v = __expf(s2[r] - nm);
      float p3v = __expf(s3[r] - nm);
      s0[r] = p0v; s1[r] = p1v; s2[r] = p2v; s3[r] = p3v;
      float ps = p0v + p1v + p2v + p3v;
      ps += __shfl_xor(ps, 1);
      ps += __shfl_xor(ps, 2);
      ps += __shfl_xor(ps, 4);
      ps += __shfl_xor(ps, 8);
      rsum[r] = rsum[r] * corr + ps;
      o0[r] *= corr; o1[r] *= corr; o2[r] *= corr; o3[r] *= corr;
    }

    // Stage P through this wave's private LDS region to get A-frag layout.
#pragma unroll
    for (int r = 0; r < 8; ++r) {
      int row = ((lane >> 4) << 3) + r;
      int col = lane & 15;
      Ps[wv][row][col]      = (bf16_t)s0[r];
      Ps[wv][row][16 + col] = (bf16_t)s1[r];
      Ps[wv][row][32 + col] = (bf16_t)s2[r];
      Ps[wv][row][48 + col] = (bf16_t)s3[r];
    }

    const int d0 = dhalf * 64;
#pragma unroll
    for (int ks = 0; ks < 2; ++ks) {
      v16bf a = load_frag_a(&Ps[wv][0][ks * 32], 64);
      o0 = wmma_bf16(a, load_frag_b_kn(&Vs[ks * 32][d0 +  0], 128), o0);
      o1 = wmma_bf16(a, load_frag_b_kn(&Vs[ks * 32][d0 + 16], 128), o1);
      o2 = wmma_bf16(a, load_frag_b_kn(&Vs[ks * 32][d0 + 32], 128), o2);
      o3 = wmma_bf16(a, load_frag_b_kn(&Vs[ks * 32][d0 + 48], 128), o3);
    }
  }

  // Normalize and scatter to [b][tok][C] bf16 (C = h*128 + d).
#pragma unroll
  for (int r = 0; r < 8; ++r) {
    float inv = 1.0f / rsum[r];
    int tok   = q0 + mi * 16 + ((lane >> 4) << 3) + r;
    int cbase = h * 128 + dhalf * 64;
    int n     = lane & 15;
    size_t base = ((size_t)(b * 1024 + tok) << 10) + cbase + n;
    attnb[base +  0] = (bf16_t)(o0[r] * inv);
    attnb[base + 16] = (bf16_t)(o1[r] * inv);
    attnb[base + 32] = (bf16_t)(o2[r] * inv);
    attnb[base + 48] = (bf16_t)(o3[r] * inv);
  }
}

// --------------------------------------------------------------- proj GEMM
// out[m][c] = sum_k A[m][k]*W[c][k] + bias[c] + resid[m][c]  (f32 out)
__global__ __launch_bounds__(256) void k_gemm_proj(const bf16_t* __restrict__ A,
                                                   const bf16_t* __restrict__ W,
                                                   const float* __restrict__ bias,
                                                   const float* __restrict__ resid,
                                                   float* __restrict__ out) {
  __shared__ __align__(16) bf16_t As[64][32];
  __shared__ __align__(16) bf16_t Bs[128][32];
  const int tid  = threadIdx.x;
  const int m0   = blockIdx.y * 64;
  const int n0   = blockIdx.x * 128;
  const int wv   = tid >> 5;
  const int lane = tid & 31;
  const int mi   = wv & 3;
  const int nj0  = (wv >> 2) * 4;

  v8f acc[4] = {};

  const int lr  = tid >> 2;
  const int lkb = (tid & 3) * 8;

  for (int k0 = 0; k0 < 1024; k0 += 32) {
#if HAVE_TDM
    if (wv == 0) {
      tdm_load_2d(&A[(size_t)m0 * 1024 + k0], lds_offset_of(&As[0][0]),
                  32, 64, 1024, 8192, 1024);
      tdm_load_2d(&W[(size_t)n0 * 1024 + k0], lds_offset_of(&Bs[0][0]),
                  32, 128, 1024, 1024, 1024);
      __builtin_amdgcn_s_wait_tensorcnt(0);
    }
#else
    *(v8bf*)&As[lr][lkb] = *(const v8bf*)&A[(size_t)(m0 + lr) * 1024 + k0 + lkb];
#pragma unroll
    for (int jj = 0; jj < 2; ++jj) {
      int n = lr + 64 * jj;
      *(v8bf*)&Bs[n][lkb] = *(const v8bf*)&W[(size_t)(n0 + n) * 1024 + k0 + lkb];
    }
    if (k0 + 32 < 1024) {
      __builtin_prefetch(&A[(size_t)(m0 + lr) * 1024 + k0 + 32 + lkb], 0, 0);
      __builtin_prefetch(&W[(size_t)(n0 + lr) * 1024 + k0 + 32 + lkb], 0, 0);
    }
#endif
    __syncthreads();
    v16bf a = load_frag_a(&As[mi * 16][0], 32);
#pragma unroll
    for (int jj = 0; jj < 4; ++jj) {
      v16bf bf = load_frag_b_nk(&Bs[(nj0 + jj) * 16][0], 32);
      acc[jj] = wmma_bf16(a, bf, acc[jj]);
    }
    __syncthreads();
  }
#pragma unroll
  for (int r = 0; r < 8; ++r) {
    int m = m0 + mi * 16 + ((lane >> 4) << 3) + r;
#pragma unroll
    for (int jj = 0; jj < 4; ++jj) {
      int c = n0 + (nj0 + jj) * 16 + (lane & 15);
      out[(size_t)m * 1024 + c] = acc[jj][r] + bias[c] + resid[(size_t)m * 1024 + c];
    }
  }
}

// ------------------------------------------------------------------ launch
extern "C" void kernel_launch(void* const* d_in, const int* in_sizes, int n_in,
                              void* d_out, int out_size, void* d_ws, size_t ws_size,
                              hipStream_t stream) {
  const float* x1     = (const float*)d_in[0];
  const float* x2     = (const float*)d_in[1];
  const float* conv_w = (const float*)d_in[2];
  const float* conv_b = (const float*)d_in[3];
  const float* wq     = (const float*)d_in[4];
  const float* wk     = (const float*)d_in[5];
  const float* wvw    = (const float*)d_in[6];
  const float* proj_w = (const float*)d_in[7];
  const float* proj_b = (const float*)d_in[8];
  float* out = (float*)d_out;

  char* ws = (char*)d_ws;
  size_t off = 0;
  auto alloc = [&](size_t bytes) -> void* {
    void* p = ws + off;
    off += (bytes + 255) & ~(size_t)255;
    return p;
  };
  const size_t ACT = (size_t)8 * 1024 * 1024;  // 8.39M elements
  bf16_t* x1b   = (bf16_t*)alloc(ACT * 2);                      // [8][1024][1024]
  bf16_t* x2b   = (bf16_t*)alloc(ACT * 2);
  bf16_t* xcatb = (bf16_t*)alloc(ACT * 2);                      // conv out
  bf16_t* wtb   = (bf16_t*)alloc((size_t)9 * 1024 * 2048 * 2);  // conv w repacked
  bf16_t* wqb   = (bf16_t*)alloc((size_t)1024 * 1024 * 2);
  bf16_t* wkb   = (bf16_t*)alloc((size_t)1024 * 1024 * 2);
  bf16_t* wvb   = (bf16_t*)alloc((size_t)1024 * 1024 * 2);
  bf16_t* wpb   = (bf16_t*)alloc((size_t)1024 * 1024 * 2);
  bf16_t* qb    = (bf16_t*)alloc(ACT * 2);                      // [b][h][tok][d]
  bf16_t* kb    = (bf16_t*)alloc(ACT * 2);
  bf16_t* vb    = (bf16_t*)alloc(ACT * 2);
  bf16_t* attnb = (bf16_t*)alloc(ACT * 2);                      // [b][tok][C]

  k_f32_to_bf16<<<1024, 256, 0, stream>>>(x1, x1b, (int)(ACT >> 3));
  k_f32_to_bf16<<<1024, 256, 0, stream>>>(x2, x2b, (int)(ACT >> 3));
  k_f32_to_bf16<<<512, 256, 0, stream>>>(wq, wqb, (1024 * 1024) >> 3);
  k_f32_to_bf16<<<512, 256, 0, stream>>>(wk, wkb, (1024 * 1024) >> 3);
  k_f32_to_bf16<<<512, 256, 0, stream>>>(wvw, wvb, (1024 * 1024) >> 3);
  k_f32_to_bf16<<<512, 256, 0, stream>>>(proj_w, wpb, (1024 * 1024) >> 3);
  k_repack_convw<<<4096, 256, 0, stream>>>(conv_w, wtb);

  k_conv<<<dim3(16, 16, 8), 256, 0, stream>>>(x1b, x2b, wtb, conv_b, xcatb);

  const float scale = 0.08838834764831845f;  // (128)^-0.5
  k_gemm_qkv<<<dim3(8, 128), 256, 0, stream>>>(x1b,   wqb, qb, scale);
  k_gemm_qkv<<<dim3(8, 128), 256, 0, stream>>>(xcatb, wkb, kb, 1.0f);
  k_gemm_qkv<<<dim3(8, 128), 256, 0, stream>>>(xcatb, wvb, vb, 1.0f);

  k_attention<<<dim3(16, 8, 8), 256, 0, stream>>>(qb, kb, vb, attnb);

  k_gemm_proj<<<dim3(8, 128), 256, 0, stream>>>(attnb, wpb, proj_b, x1, out);
}
